// MiniMindAttention_11106785427913
// MI455X (gfx1250) — matrix operand order repaired and checked by reading, not compiled, not run on hardware
//
#include <hip/hip_runtime.h>
#include <stdint.h>

// ---------------------------------------------------------------------------
// MiniMind GQA attention forward for MI455X (gfx1250, wave32, WMMA, TDM).
// B=2 S=2048 H=1024 NH=16 NKV=4 HD=64.
//
// Roofline: ~55 GFLOP total vs ~30 MB of unique data (fits in 192 MB L2,
// ~1.5 us at 23.3 TB/s) -> matrix-core bound. Everything GEMM-shaped goes
// through v_wmma_f32_16x16x32_bf16 with fp32 accumulation; softmax/RoPE in
// fp32. All tile staging is done by the Tensor Data Mover (double-buffered,
// synchronized with s_wait_tensorcnt), so the hot loops contain only
// ds_load_b128 fragment reads + WMMA. Weights and V are pre-transposed once
// (bf16) so every TDM tile is a packed 2D tile and every fragment read is
// contiguous.
// ---------------------------------------------------------------------------

typedef __bf16 bf16_t;
typedef __bf16 v16bf __attribute__((ext_vector_type(16)));
typedef float  v8f   __attribute__((ext_vector_type(8)));
typedef unsigned int u32x4v __attribute__((ext_vector_type(4)));
typedef int          i32x4v __attribute__((ext_vector_type(4)));
typedef int          i32x8v __attribute__((ext_vector_type(8)));

static constexpr int Bc  = 2;
static constexpr int Sc  = 2048;
static constexpr int Hc  = 1024;
static constexpr int NHc = 16;
static constexpr int HDc = 64;
static constexpr int ROWS = Bc * Sc;          // 4096
static constexpr int KVW  = 4 * 64;           // NKV*HD = 256

union Frag {                 // one 16x16x32 bf16 WMMA operand (16 bf16 / lane)
    v16bf  v;
    u32x4v u[2];
};

__device__ __forceinline__ v8f wmma_bf16(v16bf a, v16bf b, v8f c) {
    return __builtin_amdgcn_wmma_f32_16x16x32_bf16(false, a, false, b,
                                                   (short)0, c, false, false);
}

// ---------------------------------------------------------------------------
// Tensor Data Mover: DMA a packed 2D bf16 tile (tileW x tileH, row stride
// strideElems) from global memory into LDS at byte offset ldsOff.
// D# layout per CDNA5 ISA (data_size=1 -> 2B).
// Issued by one wave per workgroup; completion via s_wait_tensorcnt.
// ---------------------------------------------------------------------------
__device__ __forceinline__ void tdm_load_2d_bf16(uint32_t ldsOff, const void* gptr,
                                                 uint32_t tileW, uint32_t tileH,
                                                 uint32_t strideElems) {
    uint64_t ga = (uint64_t)gptr;
    u32x4v g0;
    g0[0] = 1u;                                         // count=1 (valid D#)
    g0[1] = ldsOff;                                     // lds_addr (bytes)
    g0[2] = (uint32_t)(ga & 0xFFFFFFFFu);               // global_addr[31:0]
    g0[3] = (uint32_t)((ga >> 32) & 0x01FFFFFFu) | (2u << 30);  // [56:32] | type=2

    uint64_t td0 = tileW, td1 = tileH, t0 = tileW, t1 = tileH, s0 = strideElems;
    uint64_t q0 = (1ull << 16) | ((td0 & 0xFFFFull) << 48);          // data_size=1 (2B)
    uint64_t q1 = ((td0 >> 16) & 0xFFFFull) | ((td1 & 0xFFFFFFFFull) << 16)
                | ((t0 & 0xFFFFull) << 48);
    uint64_t q2 = (t1 & 0xFFFFull) | ((s0 & 0xFFFFFFFFull) << 32);
    uint64_t q3 = ((s0 >> 32) & 0xFFFFull);             // dim1_stride unused (2D)
    i32x8v g1;
    g1[0] = (int)(uint32_t)q0;  g1[1] = (int)(uint32_t)(q0 >> 32);
    g1[2] = (int)(uint32_t)q1;  g1[3] = (int)(uint32_t)(q1 >> 32);
    g1[4] = (int)(uint32_t)q2;  g1[5] = (int)(uint32_t)(q2 >> 32);
    g1[6] = (int)(uint32_t)q3;  g1[7] = (int)(uint32_t)(q3 >> 32);
    i32x4v g2 = {0, 0, 0, 0};
    i32x4v g3 = {0, 0, 0, 0};
#if defined(__clang_major__) && __clang_major__ >= 23
    i32x8v g4 = {0, 0, 0, 0, 0, 0, 0, 0};
    __builtin_amdgcn_tensor_load_to_lds(g0, g1, g2, g3, g4, 0);
#else
    __builtin_amdgcn_tensor_load_to_lds(g0, g1, g2, g3, 0);
#endif
}

// ---------------------------------------------------------------------------
// fp32 -> bf16 (x only; weights go through the transposing converters)
// ---------------------------------------------------------------------------
__global__ void cvt_f32_bf16(const float* __restrict__ src, bf16_t* __restrict__ dst, int n) {
    int i = blockIdx.x * blockDim.x + threadIdx.x;
    if (i < n) dst[i] = (bf16_t)src[i];
}

// ---------------------------------------------------------------------------
// Transpose-convert a K x N fp32 weight into N x K bf16 (LDS-tiled, coalesced
// on both sides).  Grid = (K/32, N/32), 256 threads.
// ---------------------------------------------------------------------------
__global__ void __launch_bounds__(256)
transpose_w_f32_bf16(const float* __restrict__ src, bf16_t* __restrict__ dst,
                     int K, int N) {
    __shared__ float t[32][33];
    int k0 = blockIdx.x * 32, n0 = blockIdx.y * 32;
    int tx = threadIdx.x & 31, ty = threadIdx.x >> 5;
#pragma unroll
    for (int i = ty; i < 32; i += 8)
        t[i][tx] = src[(size_t)(k0 + i) * N + n0 + tx];
    __syncthreads();
#pragma unroll
    for (int i = ty; i < 32; i += 8)
        dst[(size_t)(n0 + i) * K + k0 + tx] = (bf16_t)t[tx][i];
}

// ---------------------------------------------------------------------------
// V slices: (B*S, NKV*HD) fp32 -> Vt (B, NKV, HD, S) bf16, one (b,g) slice
// per blockIdx.z.  Grid = (S/32, HD/32, B*NKV).
// ---------------------------------------------------------------------------
__global__ void __launch_bounds__(256)
transpose_v_f32_bf16(const float* __restrict__ v, bf16_t* __restrict__ vt) {
    __shared__ float t[32][33];
    int z = blockIdx.z, b = z >> 2, g = z & 3;
    const float* s = v + (size_t)b * Sc * KVW + g * HDc;
    bf16_t* d = vt + (size_t)z * HDc * Sc;
    int r0 = blockIdx.x * 32, c0 = blockIdx.y * 32;
    int tx = threadIdx.x & 31, ty = threadIdx.x >> 5;
#pragma unroll
    for (int i = ty; i < 32; i += 8)
        t[i][tx] = s[(size_t)(r0 + i) * KVW + c0 + tx];
    __syncthreads();
#pragma unroll
    for (int i = ty; i < 32; i += 8)
        d[(size_t)(c0 + i) * Sc + r0 + tx] = (bf16_t)t[tx][i];
}

// ---------------------------------------------------------------------------
// RoPE (rotate-half) fp32 -> bf16.  width = heads*HD of this tensor.
// ---------------------------------------------------------------------------
__global__ void rope_f32_bf16(const float* __restrict__ src, const float* __restrict__ cosT,
                              const float* __restrict__ sinT, bf16_t* __restrict__ dst,
                              int width, int n) {
    int i = blockIdx.x * blockDim.x + threadIdx.x;
    if (i >= n) return;
    int col = i % width;
    int s   = (i / width) % Sc;
    int d   = col & (HDc - 1);
    float x  = src[i];
    float c  = cosT[s * HDc + d];
    float sn = sinT[s * HDc + d];
    float other = (d < HDc / 2) ? -src[i + HDc / 2] : src[i - HDc / 2];
    dst[i] = (bf16_t)(x * c + other * sn);
}

// ---------------------------------------------------------------------------
// bf16 WMMA GEMM: C(f32, MxN) = A(bf16, MxK row-major) * Bt(bf16, NxK
// row-major = B^T).  256 threads = 8 waves (4x2), wave tile 32x64, WG tile
// 128x128.  Both tiles DMA'd by the TDM into double-buffered LDS; hot loop is
// pure ds_load_b128 + v_wmma.
//   LDS: buf b at b*16KB:  As[128][32] (8KB) | Bs[128][32] (8KB)
// ---------------------------------------------------------------------------
__global__ void __launch_bounds__(256)
gemm_bf16_wmma(const bf16_t* __restrict__ A, const bf16_t* __restrict__ Bt,
               float* __restrict__ C, int M, int N, int K) {
    extern __shared__ char smem[];
    const int tid  = threadIdx.x;
    const int lane = tid & 31, wid = tid >> 5;
    const int wm = wid >> 1, wn = wid & 1;      // 4x2 wave grid
    const int bm = blockIdx.x * 128, bn = blockIdx.y * 128;
    const int half = lane >> 4, l16 = lane & 15;

    const bf16_t* Abase = A  + (size_t)bm * K;
    const bf16_t* Bbase = Bt + (size_t)bn * K;

    v8f acc[2][4] = {};

    if (wid == 0) {                              // prime buffer 0
        tdm_load_2d_bf16(0u,    Abase, 32u, 128u, (uint32_t)K);
        tdm_load_2d_bf16(8192u, Bbase, 32u, 128u, (uint32_t)K);
        __builtin_amdgcn_s_wait_tensorcnt(0);
    }
    __syncthreads();

    for (int k0 = 0; k0 < K; k0 += 32) {
        const int cur = (k0 >> 5) & 1, nxt = cur ^ 1;
        if (k0 + 32 < K && wid == 0) {           // prefetch next buffer (DMA
            tdm_load_2d_bf16(nxt * 16384u,         Abase + k0 + 32, 32u, 128u, (uint32_t)K);
            tdm_load_2d_bf16(nxt * 16384u + 8192u, Bbase + k0 + 32, 32u, 128u, (uint32_t)K);
        }                                        //  overlaps WMMA below)
        const bf16_t* As = (const bf16_t*)(smem + cur * 16384);
        const bf16_t* Bs = (const bf16_t*)(smem + cur * 16384 + 8192);

        Frag aF[2], bF[4];
#pragma unroll
        for (int mt = 0; mt < 2; ++mt) {   // A frag: M=l16, K=16*(j>>3)+8*half+(j&7)
            const bf16_t* ar = As + (wm * 32 + mt * 16 + l16) * 32;
            aF[mt].u[0] = *(const u32x4v*)(ar + half * 8);
            aF[mt].u[1] = *(const u32x4v*)(ar + 16 + half * 8);
        }
#pragma unroll
        for (int nt = 0; nt < 4; ++nt) {   // B frag: N=l16, K=16*(j>>3)+8*half+(j&7)
            const bf16_t* br = Bs + (wn * 64 + nt * 16 + l16) * 32;
            bF[nt].u[0] = *(const u32x4v*)(br + half * 8);
            bF[nt].u[1] = *(const u32x4v*)(br + 16 + half * 8);
        }
#pragma unroll
        for (int mt = 0; mt < 2; ++mt)
#pragma unroll
            for (int nt = 0; nt < 4; ++nt)
                acc[mt][nt] = wmma_bf16(aF[mt].v, bF[nt].v, acc[mt][nt]);

        if (wid == 0) __builtin_amdgcn_s_wait_tensorcnt(0);
        __syncthreads();
    }

    // C/D layout: row = r + 8*half, col = l16
#pragma unroll
    for (int mt = 0; mt < 2; ++mt)
#pragma unroll
        for (int nt = 0; nt < 4; ++nt)
#pragma unroll
            for (int r = 0; r < 8; ++r) {
                int row = bm + wm * 32 + mt * 16 + r + 8 * half;
                int col = bn + wn * 64 + nt * 16 + l16;
                C[(size_t)row * N + col] = acc[mt][nt][r];
            }
}

// ---------------------------------------------------------------------------
// Flash attention (causal GQA).  Grid = (S/128, NH, B), 128 threads = 4 waves,
// each wave owns 32 query rows.  Q tile TDM'd once; K and V^T tiles TDM'd per
// 32-key block, double-buffered.  Per block per wave: 8 WMMA (QK^T) + 8 WMMA
// (PV).  Online softmax in fp32; P converted via per-wave LDS round-trip.
//   LDS: Q[128][64] 16KB | {K[32][64] 4KB, Vt[64][32] 4KB} x2 | P 4x[32][32] 8KB
// ---------------------------------------------------------------------------
__global__ void __launch_bounds__(128)
flash_attn_wmma(const bf16_t* __restrict__ Q,   // (B*S, NH*HD) bf16 post-RoPE
                const bf16_t* __restrict__ Kb,  // (B*S, NKV*HD) bf16 post-RoPE
                const bf16_t* __restrict__ Vt,  // (B, NKV, HD, S) bf16
                const float*  __restrict__ am,  // (B, S)
                bf16_t* __restrict__ O) {       // (B*S, NH*HD) bf16
    extern __shared__ char smem[];
    const int tid  = threadIdx.x;
    const int lane = tid & 31, wid = tid >> 5;
    const int half = lane >> 4, l16 = lane & 15;
    const int qb = blockIdx.x, h = blockIdx.y, b = blockIdx.z;
    const int g  = h >> 2;                       // kv group = h / R, R=4
    const int q0 = qb * 128;

    const bf16_t* Qg = Q  + (size_t)(b * Sc + q0) * (NHc * HDc) + h * HDc;
    const bf16_t* Kg = Kb + (size_t)(b * Sc) * KVW + g * HDc;
    const bf16_t* Vg = Vt + (size_t)(b * 4 + g) * HDc * Sc;

    if (wid == 0) {                              // Q tile + first K/V tiles
        tdm_load_2d_bf16(0u,     Qg, 64u, 128u, (uint32_t)(NHc * HDc));
        tdm_load_2d_bf16(16384u, Kg, 64u, 32u,  (uint32_t)KVW);
        tdm_load_2d_bf16(20480u, Vg, 32u, 64u,  (uint32_t)Sc);
        __builtin_amdgcn_s_wait_tensorcnt(0);
    }
    __syncthreads();

    // Q fragments live in registers for the whole kernel
    const bf16_t* ldsQ = (const bf16_t*)smem;
    Frag qF[2][2];
#pragma unroll
    for (int mt = 0; mt < 2; ++mt)
#pragma unroll
        for (int ks = 0; ks < 2; ++ks) {
            const bf16_t* ar = ldsQ + (wid * 32 + mt * 16 + l16) * HDc + ks * 32;
            qF[mt][ks].u[0] = *(const u32x4v*)(ar + half * 8);
            qF[mt][ks].u[1] = *(const u32x4v*)(ar + 16 + half * 8);
        }

    float m_i[2][8], l_i[2][8];
    v8f o[2][4] = {};
#pragma unroll
    for (int mt = 0; mt < 2; ++mt)
#pragma unroll
        for (int r = 0; r < 8; ++r) { m_i[mt][r] = -1e30f; l_i[mt][r] = 0.0f; }

    bf16_t* ldsP = (bf16_t*)(smem + 32768) + wid * 1024;   // per-wave 32x32

    const int kbMax = (q0 + 127) >> 5;
    for (int kb = 0; kb <= kbMax; ++kb) {
        const int cur = kb & 1, nxt = cur ^ 1;
        if (kb < kbMax && wid == 0) {            // prefetch next K/V tiles
            tdm_load_2d_bf16(16384u + nxt * 8192u,
                             Kg + (size_t)(kb + 1) * 32 * KVW, 64u, 32u, (uint32_t)KVW);
            tdm_load_2d_bf16(20480u + nxt * 8192u,
                             Vg + (kb + 1) * 32, 32u, 64u, (uint32_t)Sc);
        }
        const bf16_t* ldsK  = (const bf16_t*)(smem + 16384 + cur * 8192);
        const bf16_t* ldsVT = (const bf16_t*)(smem + 20480 + cur * 8192);

        // ---- S = Q K^T  (contraction over HD=64: two K=32 WMMA steps) ----
        v8f s[2][2] = {};
#pragma unroll
        for (int nt = 0; nt < 2; ++nt) {
            Frag kF0, kF1;                        // B frag: N=key, K=hd
            const bf16_t* kr = ldsK + (nt * 16 + l16) * HDc + half * 8;
            kF0.u[0] = *(const u32x4v*)(kr);      kF0.u[1] = *(const u32x4v*)(kr + 16);
            kF1.u[0] = *(const u32x4v*)(kr + 32); kF1.u[1] = *(const u32x4v*)(kr + 48);
#pragma unroll
            for (int mt = 0; mt < 2; ++mt) {
                s[mt][nt] = wmma_bf16(qF[mt][0].v, kF0.v, s[mt][nt]);
                s[mt][nt] = wmma_bf16(qF[mt][1].v, kF1.v, s[mt][nt]);
            }
        }

        // ---- mask + online softmax (fp32) ----
        int   kcol[2];
        float amt[2];
#pragma unroll
        for (int nt = 0; nt < 2; ++nt) {
            kcol[nt] = kb * 32 + nt * 16 + l16;
            amt[nt]  = (1.0f - am[b * Sc + kcol[nt]]) * -1e9f;
        }
#pragma unroll
        for (int mt = 0; mt < 2; ++mt) {
            float rmax[8];
#pragma unroll
            for (int r = 0; r < 8; ++r) {
                int qrow = q0 + wid * 32 + mt * 16 + r + 8 * half;
                float best = -1e30f;
#pragma unroll
                for (int nt = 0; nt < 2; ++nt) {
                    float sc = s[mt][nt][r] * 0.125f + amt[nt];   // 1/sqrt(HD)
                    if (kcol[nt] > qrow) sc = -1e30f;             // causal
                    s[mt][nt][r] = sc;
                    best = fmaxf(best, sc);
                }
#pragma unroll
                for (int d = 8; d >= 1; d >>= 1)   // row-reduce over 16 lanes
                    best = fmaxf(best, __shfl_xor(best, d, 32));
                rmax[r] = best;
            }
            float cexp[8];
#pragma unroll
            for (int r = 0; r < 8; ++r) {
                float mnew = fmaxf(m_i[mt][r], rmax[r]);
                cexp[r] = __expf(m_i[mt][r] - mnew);
                m_i[mt][r] = mnew;
            }
#pragma unroll
            for (int r = 0; r < 8; ++r) {
                float psum = 0.0f;
#pragma unroll
                for (int nt = 0; nt < 2; ++nt) {
                    float p = __expf(s[mt][nt][r] - m_i[mt][r]);
                    s[mt][nt][r] = p;
                    psum += p;
                }
#pragma unroll
                for (int d = 8; d >= 1; d >>= 1)
                    psum += __shfl_xor(psum, d, 32);
                l_i[mt][r] = l_i[mt][r] * cexp[r] + psum;
            }
#pragma unroll
            for (int nto = 0; nto < 4; ++nto)
#pragma unroll
                for (int r = 0; r < 8; ++r) o[mt][nto][r] *= cexp[r];

            // P (C-layout) -> per-wave LDS, re-read below in A-fragment layout
#pragma unroll
            for (int nt = 0; nt < 2; ++nt)
#pragma unroll
                for (int r = 0; r < 8; ++r)
                    ldsP[(mt * 16 + r + 8 * half) * 32 + nt * 16 + l16] =
                        (bf16_t)s[mt][nt][r];
        }
        asm volatile("s_wait_dscnt 0" ::: "memory");   // own-wave P visibility

        // ---- O += P V  (contraction over 32 keys) ----
#pragma unroll
        for (int mt = 0; mt < 2; ++mt) {
            Frag pF;                                   // A frag of P
            const bf16_t* pr = ldsP + (mt * 16 + l16) * 32;
            pF.u[0] = *(const u32x4v*)(pr + half * 8);
            pF.u[1] = *(const u32x4v*)(pr + 16 + half * 8);
#pragma unroll
            for (int nto = 0; nto < 4; ++nto) {
                Frag vF;                               // B frag: N=hd, K=key
                const bf16_t* vr = ldsVT + (nto * 16 + l16) * 32 + half * 8;
                vF.u[0] = *(const u32x4v*)(vr);
                vF.u[1] = *(const u32x4v*)(vr + 16);
                o[mt][nto] = wmma_bf16(pF.v, vF.v, o[mt][nto]);
            }
        }

        if (wid == 0) __builtin_amdgcn_s_wait_tensorcnt(0);
        __syncthreads();
    }

    // ---- normalize and store bf16 attention output ----
#pragma unroll
    for (int mt = 0; mt < 2; ++mt)
#pragma unroll
        for (int r = 0; r < 8; ++r) {
            float invl = 1.0f / l_i[mt][r];
            int qrow = q0 + wid * 32 + mt * 16 + r + 8 * half;
#pragma unroll
            for (int nto = 0; nto < 4; ++nto)
                O[(size_t)(b * Sc + qrow) * (NHc * HDc) + h * HDc + nto * 16 + l16] =
                    (bf16_t)(o[mt][nto][r] * invl);
        }
}

// ---------------------------------------------------------------------------
// Launcher
// ---------------------------------------------------------------------------
extern "C" void kernel_launch(void* const* d_in, const int* in_sizes, int n_in,
                              void* d_out, int out_size, void* d_ws, size_t ws_size,
                              hipStream_t stream) {
    const float* x    = (const float*)d_in[0];
    const float* cosT = (const float*)d_in[1];
    const float* sinT = (const float*)d_in[2];
    const float* amsk = (const float*)d_in[3];
    const float* Wq   = (const float*)d_in[4];
    const float* Wk   = (const float*)d_in[5];
    const float* Wv   = (const float*)d_in[6];
    const float* Wo   = (const float*)d_in[7];
    float* out = (float*)d_out;

    char* ws = (char*)d_ws;
    size_t off = 0;
    auto alloc = [&](size_t bytes) -> char* {
        char* p = ws + off;
        off += (bytes + 255) & ~(size_t)255;
        return p;
    };
    bf16_t* xb   = (bf16_t*)alloc((size_t)ROWS * Hc * 2);   // x bf16 (A matrix)
    bf16_t* wqT  = (bf16_t*)alloc((size_t)Hc * Hc * 2);     // Wq^T bf16 (NxK)
    bf16_t* wkT  = (bf16_t*)alloc((size_t)Hc * KVW * 2);
    bf16_t* wvT  = (bf16_t*)alloc((size_t)Hc * KVW * 2);
    bf16_t* woT  = (bf16_t*)alloc((size_t)Hc * Hc * 2);
    float*  qf   = (float*) alloc((size_t)ROWS * Hc * 4);   // Q pre-RoPE
    float*  kf   = (float*) alloc((size_t)ROWS * KVW * 4);
    float*  vf   = (float*) alloc((size_t)ROWS * KVW * 4);
    bf16_t* qbb  = (bf16_t*)alloc((size_t)ROWS * Hc * 2);   // Q post-RoPE bf16
    bf16_t* kbb  = (bf16_t*)alloc((size_t)ROWS * KVW * 2);
    bf16_t* vtb  = (bf16_t*)alloc((size_t)ROWS * KVW * 2);  // V^T (B,NKV,HD,S)
    bf16_t* atb  = (bf16_t*)alloc((size_t)ROWS * Hc * 2);   // attention output

    auto cdiv = [](int a, int b) { return (a + b - 1) / b; };

    // 1) fp32 -> bf16 (x), fp32 -> bf16 transposed (weights)
    cvt_f32_bf16<<<cdiv(ROWS * Hc, 256), 256, 0, stream>>>(x, xb, ROWS * Hc);
    transpose_w_f32_bf16<<<dim3(Hc / 32, Hc / 32),  256, 0, stream>>>(Wq, wqT, Hc, Hc);
    transpose_w_f32_bf16<<<dim3(Hc / 32, KVW / 32), 256, 0, stream>>>(Wk, wkT, Hc, KVW);
    transpose_w_f32_bf16<<<dim3(Hc / 32, KVW / 32), 256, 0, stream>>>(Wv, wvT, Hc, KVW);
    transpose_w_f32_bf16<<<dim3(Hc / 32, Hc / 32),  256, 0, stream>>>(Wo, woT, Hc, Hc);

    // 2) QKV projections (TDM-staged bf16 WMMA, fp32 accumulate)
    const size_t gemmLds = 2 * 16384;                 // double-buffered tiles
    dim3 gq (ROWS / 128, Hc  / 128);
    dim3 gkv(ROWS / 128, KVW / 128);
    gemm_bf16_wmma<<<gq,  256, gemmLds, stream>>>(xb, wqT, qf, ROWS, Hc,  Hc);
    gemm_bf16_wmma<<<gkv, 256, gemmLds, stream>>>(xb, wkT, kf, ROWS, KVW, Hc);
    gemm_bf16_wmma<<<gkv, 256, gemmLds, stream>>>(xb, wvT, vf, ROWS, KVW, Hc);

    // 3) RoPE (fp32) -> bf16; V transposed to (B,NKV,HD,S) bf16
    rope_f32_bf16<<<cdiv(ROWS * Hc, 256),  256, 0, stream>>>(qf, cosT, sinT, qbb, Hc,  ROWS * Hc);
    rope_f32_bf16<<<cdiv(ROWS * KVW, 256), 256, 0, stream>>>(kf, cosT, sinT, kbb, KVW, ROWS * KVW);
    transpose_v_f32_bf16<<<dim3(Sc / 32, HDc / 32, Bc * 4), 256, 0, stream>>>(vf, vtb);

    // 4) causal GQA flash attention (TDM-staged, double-buffered K/V)
    dim3 ga(Sc / 128, NHc, Bc);
    flash_attn_wmma<<<ga, 128, 40960, stream>>>(qbb, kbb, vtb, amsk, atb);

    // 5) output projection -> fp32 d_out
    gemm_bf16_wmma<<<gq, 256, gemmLds, stream>>>(atb, woT, out, ROWS, Hc, Hc);
}
